// GradGNN_46377056862937
// MI455X (gfx1250) — matrix operand order repaired and checked by reading, not compiled, not run on hardware
//
#include <hip/hip_runtime.h>
#include <hip/hip_bf16.h>

// GradGNN on MI455X (gfx1250):
//  - 8 SpMM passes are L2-resident (Tx buffers 51MB << 192MB L2); scatter uses
//    native global_atomic_add_f32 (inline asm, avoids CAS expansion).
//  - All dense GEMMs use v_wmma_f32_16x16x32_bf16 with f32 accumulation.
//  - Weights are pre-transposed to bf16 W^T once (tiny, L2-resident); each GEMM
//    block stages its W^T tile into LDS with the Tensor Data Mover
//    (tensor_load_to_lds + s_wait_tensorcnt), using D# pad fields to produce the
//    128->136-element padded rows that make B-fragment ds_load_b128s bank-spread.

#define IN_DIM_ 128
#define HID_    128
#define OUT_DIM_ 64
#define ORDER_    4

typedef __attribute__((ext_vector_type(16))) __bf16   v16bf;
typedef __attribute__((ext_vector_type(8)))  __bf16   v8bf;
typedef __attribute__((ext_vector_type(8)))  float    v8f;
typedef __attribute__((ext_vector_type(4)))  unsigned v4u;
typedef __attribute__((ext_vector_type(8)))  int      v8i_;
typedef __attribute__((ext_vector_type(4)))  int      v4i_;

__device__ __forceinline__ void gatomic_add_f32(float* p, float v) {
  // CDNA5 GLOBAL_ATOMIC_ADD_F32 (no return) — tracked via STOREcnt, drained at s_endpgm.
  asm volatile("global_atomic_add_f32 %0, %1, off" :: "v"(p), "v"(v) : "memory");
}

// ---- TDM: stage `rows` rows of 256B (64 DWORDs) from global bf16 W^T into LDS,
// inserting 16B (4 DWORDs) of padding after every row (row stride 272B = 136 bf16).
__device__ __forceinline__ void tdm_stage_wt(const __bf16* gsrc, __bf16* lds_dst, int rows) {
  unsigned long long ga = (unsigned long long)(size_t)gsrc;   // global byte address
  v4u g0;
  g0[0] = 1u;                                  // count=1 (valid user descriptor)
  g0[1] = (unsigned)(size_t)lds_dst;           // LDS byte offset (flat addr low 32 bits)
  g0[2] = (unsigned)ga;                        // global_addr[31:0]
  g0[3] = (unsigned)(ga >> 32) | (2u << 30);   // global_addr[56:32] | type=2 ("image")
  v8i_ g1;
  g1[0] = (int)((2u << 16)      // data_size = 4B
              | (1u << 20)      // pad_enable
              | (5u << 22)      // pad_interval: every 64 DWORDs
              | (3u << 25));    // pad_amount: 4 DWORDs (16B)
  g1[1] = (int)(64u << 16);     // tensor_dim0 = 64 (DWORDs per row), low 16 bits
  g1[2] = (int)((unsigned)rows << 16); // tensor_dim1 = rows (low 16 bits)
  g1[3] = (int)(64u << 16);     // tile_dim0 = 64
  g1[4] = rows;                 // tile_dim1 = rows (tile_dim2 = 0)
  g1[5] = 64;                   // tensor_dim0_stride = 64 DWORDs (contiguous rows)
  g1[6] = 0;                    // tensor_dim1_stride unused for 2D tile
  g1[7] = 0;
  v4i_ z4 = {0, 0, 0, 0};               // groups 2/3: no dims 2..4
  v8i_ z8 = {0, 0, 0, 0, 0, 0, 0, 0};
  __builtin_amdgcn_tensor_load_to_lds(g0, g1, z4, z4, z8, 0);
  __builtin_amdgcn_s_wait_tensorcnt(0);
}

// ---------------- small utility kernels ----------------
__global__ void k_zero_u32(unsigned* __restrict__ p, int n) {
  int i = blockIdx.x * blockDim.x + threadIdx.x;
  if (i < n) p[i] = 0u;
}

__global__ void k_deg(const long long* __restrict__ cols, unsigned* __restrict__ deg, int ne) {
  int i = blockIdx.x * blockDim.x + threadIdx.x;
  if (i < ne) atomicAdd(&deg[(int)cols[i]], 1u);
}

__global__ void k_dinv(const unsigned* __restrict__ deg, float* __restrict__ dinv, int n) {
  int i = blockIdx.x * blockDim.x + threadIdx.x;
  if (i < n) dinv[i] = rsqrtf((float)deg[i] + 1.0f);  // +1 = self loop; always > 0
}

// W^T bf16 materialization: out[c*K + k] = (bf16)W[k*C + c]
__global__ void k_wt_bf16(const float* __restrict__ W, __bf16* __restrict__ out, int K, int C) {
  int i = blockIdx.x * blockDim.x + threadIdx.x;
  if (i >= K * C) return;
  int k = i / C, c = i % C;
  out[(size_t)c * K + k] = (__bf16)W[i];
}

// dst[n] = dinv[n]^2 * src[n]  (self-loop term initializes the scatter target)
__global__ void k_selfloop(float* __restrict__ dst, const float* __restrict__ src,
                           const float* __restrict__ dinv, int n) {
  int t = blockIdx.x * blockDim.x + threadIdx.x;
  int node = t >> 5, lane = t & 31;
  if (node >= n) return;
  float s = dinv[node]; s = s * s;
  const float4 v = *(const float4*)(src + (size_t)node * HID_ + lane * 4);
  float4 o; o.x = s * v.x; o.y = s * v.y; o.z = s * v.z; o.w = s * v.w;
  *(float4*)(dst + (size_t)node * HID_ + lane * 4) = o;
}

// one wave (32 lanes) per edge; each lane moves 4 floats (float4 gather, 4 atomics)
__global__ void k_scatter(const long long* __restrict__ rows, const long long* __restrict__ cols,
                          const float* __restrict__ dinv, const float* __restrict__ src,
                          float* __restrict__ dst, int ne) {
  int t = blockIdx.x * blockDim.x + threadIdx.x;
  int e = t >> 5, lane = t & 31;
  if (e >= ne) return;
  int r = (int)rows[e], c = (int)cols[e];          // uniform across the wave
  float nrm = dinv[r] * dinv[c];
  const float4 v = *(const float4*)(src + (size_t)r * HID_ + lane * 4);
  float* d = dst + (size_t)c * HID_ + lane * 4;
  gatomic_add_f32(d + 0, nrm * v.x);
  gatomic_add_f32(d + 1, nrm * v.y);
  gatomic_add_f32(d + 2, nrm * v.z);
  gatomic_add_f32(d + 3, nrm * v.w);
}

// ---------------- WMMA GEMM: out = (init? bias : out+bias) + A @ W, optional relu ----------------
// A: [nrows][128] f32; wT: [128][128] bf16 (pre-transposed W); out: [nrows][128] f32.
// Per wave: 16-row tile x 8 col tiles, K=128 in 4 chunks of 32 (bf16 WMMA, f32 acc).
__global__ __launch_bounds__(256) void k_gemm128_acc(
    const float* __restrict__ A, const __bf16* __restrict__ wT,
    const float* __restrict__ bias, float* __restrict__ out,
    int nrows, int init, int do_relu)
{
  __shared__ alignas(16) __bf16 wt[HID_ * 136];   // W^T, rows padded 128->136 (bank spread)
  if (threadIdx.x < 32) tdm_stage_wt(wT, wt, HID_);
  __syncthreads();

  const int wave = threadIdx.x >> 5, lane = threadIdx.x & 31;
  const int m = lane & 15, half = lane >> 4;
  const long long row0 = ((long long)blockIdx.x * 8 + wave) * 16;
  if (row0 >= nrows) return;                      // wave-uniform: EXEC stays all-ones

  v8f acc[8];
#pragma unroll
  for (int n = 0; n < 8; ++n) {
    float bv = bias[n * 16 + m];
    const float* p = out + (row0 + half * 8) * HID_ + n * 16 + m;
#pragma unroll
    for (int r = 0; r < 8; ++r)
      acc[n][r] = bv + (init ? 0.0f : p[r * HID_]);
  }

  const float* arow = A + (row0 + m) * (long long)IN_DIM_;
#pragma unroll
  for (int kb = 0; kb < 4; ++kb) {
    const int k0 = kb * 32 + half * 8;            // 16-bit A layout: lanes<16 K{0..7,16..23}, else {8..15,24..31}
    float4 a0 = *(const float4*)(arow + k0);
    float4 a1 = *(const float4*)(arow + k0 + 4);
    float4 a2 = *(const float4*)(arow + k0 + 16);
    float4 a3 = *(const float4*)(arow + k0 + 20);
    v16bf a;
    a[0]=(__bf16)a0.x; a[1]=(__bf16)a0.y; a[2]=(__bf16)a0.z; a[3]=(__bf16)a0.w;
    a[4]=(__bf16)a1.x; a[5]=(__bf16)a1.y; a[6]=(__bf16)a1.z; a[7]=(__bf16)a1.w;
    a[8]=(__bf16)a2.x; a[9]=(__bf16)a2.y; a[10]=(__bf16)a2.z; a[11]=(__bf16)a2.w;
    a[12]=(__bf16)a3.x; a[13]=(__bf16)a3.y; a[14]=(__bf16)a3.z; a[15]=(__bf16)a3.w;
    // Preload all B fragments for this K-chunk, then back-to-back WMMAs.
    v16bf bfrag[8];
#pragma unroll
    for (int n = 0; n < 8; ++n) {
      // B layout: lane holds col N=lane%16; k_local = 8*half+v (v<8) or 16+8*half+(v-8), matching A
      const __bf16* bp = &wt[(n * 16 + m) * 136 + kb * 32 + half * 8];
      v8bf blo = *(const v8bf*)bp;
      v8bf bhi = *(const v8bf*)(bp + 16);
      bfrag[n] = __builtin_shufflevector(blo, bhi, 0,1,2,3,4,5,6,7,8,9,10,11,12,13,14,15);
    }
#pragma unroll
    for (int n = 0; n < 8; ++n)
      acc[n] = __builtin_amdgcn_wmma_f32_16x16x32_bf16(false, a, false, bfrag[n], (short)0, acc[n], false, false);
  }

#pragma unroll
  for (int n = 0; n < 8; ++n) {
    float* p = out + (row0 + half * 8) * HID_ + n * 16 + m;
#pragma unroll
    for (int r = 0; r < 8; ++r) {
      float v = acc[n][r];
      p[r * HID_] = do_relu ? fmaxf(v, 0.0f) : v;
    }
  }
}

// ---------------- fusion: h = softmax(fw)[0]*o0 + [1]*o1, stored bf16 ----------------
__global__ void k_fuse(const float* __restrict__ o0, const float* __restrict__ o1,
                       const float* __restrict__ fw, __bf16* __restrict__ h, long long total) {
  long long i = (long long)blockIdx.x * blockDim.x + threadIdx.x;
  if (i >= total) return;
  float a = fw[0], b = fw[1];
  float mx = fmaxf(a, b);
  float ea = __expf(a - mx), eb = __expf(b - mx);
  float inv = 1.0f / (ea + eb);
  h[i] = (__bf16)((ea * inv) * o0[i] + (eb * inv) * o1[i]);
}

// ---------------- classifier WMMA GEMM: out[N][64] = h @ clf_w + clf_b ----------------
__global__ __launch_bounds__(256) void k_clf(
    const __bf16* __restrict__ A, const __bf16* __restrict__ wT,   // wT: [64][128] bf16
    const float* __restrict__ bias, float* __restrict__ out, int nrows)
{
  __shared__ alignas(16) __bf16 wt[OUT_DIM_ * 136];
  if (threadIdx.x < 32) tdm_stage_wt(wT, wt, OUT_DIM_);
  __syncthreads();

  const int wave = threadIdx.x >> 5, lane = threadIdx.x & 31;
  const int m = lane & 15, half = lane >> 4;
  const long long row0 = ((long long)blockIdx.x * 8 + wave) * 16;
  if (row0 >= nrows) return;

  v8f acc[4];
#pragma unroll
  for (int n = 0; n < 4; ++n) {
    float bv = bias[n * 16 + m];
#pragma unroll
    for (int r = 0; r < 8; ++r) acc[n][r] = bv;
  }

  const __bf16* arow = A + (row0 + m) * (long long)HID_;
#pragma unroll
  for (int kb = 0; kb < 4; ++kb) {
    const int k0 = kb * 32 + half * 8;
    v8bf alo = *(const v8bf*)(arow + k0);
    v8bf ahi = *(const v8bf*)(arow + k0 + 16);
    v16bf a = __builtin_shufflevector(alo, ahi, 0,1,2,3,4,5,6,7,8,9,10,11,12,13,14,15);
    v16bf bfrag[4];
#pragma unroll
    for (int n = 0; n < 4; ++n) {
      const __bf16* bp = &wt[(n * 16 + m) * 136 + kb * 32 + half * 8];
      v8bf blo = *(const v8bf*)bp;
      v8bf bhi = *(const v8bf*)(bp + 16);
      bfrag[n] = __builtin_shufflevector(blo, bhi, 0,1,2,3,4,5,6,7,8,9,10,11,12,13,14,15);
    }
#pragma unroll
    for (int n = 0; n < 4; ++n)
      acc[n] = __builtin_amdgcn_wmma_f32_16x16x32_bf16(false, a, false, bfrag[n], (short)0, acc[n], false, false);
  }

#pragma unroll
  for (int n = 0; n < 4; ++n) {
    float* p = out + (row0 + half * 8) * OUT_DIM_ + n * 16 + m;
#pragma unroll
    for (int r = 0; r < 8; ++r) p[r * OUT_DIM_] = acc[n][r];
  }
}

// ---------------- host orchestration ----------------
extern "C" void kernel_launch(void* const* d_in, const int* in_sizes, int n_in,
                              void* d_out, int out_size, void* d_ws, size_t ws_size,
                              hipStream_t stream) {
  (void)n_in; (void)out_size; (void)ws_size;
  const float*     x      = (const float*)d_in[0];
  const long long* ei     = (const long long*)d_in[1];   // [2][E] int64
  const long long* gei    = (const long long*)d_in[2];
  const float*     w_orig = (const float*)d_in[3];       // [5][128][128]
  const float*     b_orig = (const float*)d_in[4];       // [5][128]
  const float*     w_aug  = (const float*)d_in[5];
  const float*     b_aug  = (const float*)d_in[6];
  const float*     fw     = (const float*)d_in[7];       // [2]
  const float*     clf_w  = (const float*)d_in[8];       // [128][64]
  const float*     clf_b  = (const float*)d_in[9];       // [64]

  const int N = in_sizes[0] / IN_DIM_;
  const int E = in_sizes[1] / 2;
  const size_t NF = (size_t)N * HID_ * sizeof(float);
  const size_t MATE = (size_t)IN_DIM_ * HID_;            // 16384 elements per weight matrix

  char* p = (char*)d_ws;
  float*    out0 = (float*)p;    p += NF;
  float*    out1 = (float*)p;    p += NF;
  float*    txA  = (float*)p;    p += NF;
  float*    txB  = (float*)p;    p += NF;
  float*    dinv = (float*)p;    p += (size_t)N * sizeof(float);
  unsigned* deg  = (unsigned*)p; p += (size_t)N * sizeof(unsigned);
  __bf16*   wT   = (__bf16*)p;   p += (10 * MATE + (size_t)HID_ * OUT_DIM_) * sizeof(__bf16);
  __bf16*   hbuf = (__bf16*)txA; // txA is dead by fusion time

  const int T = 256;
  const int rowTiles   = (N + 15) / 16;
  const int gemmBlocks = (rowTiles + 7) / 8;
  const int nodeBlocks = (N + T - 1) / T;
  const int edgeBlocks = (E + T - 1) / T;
  const int slBlocks   = (int)(((long long)N * 32 + T - 1) / T);
  const int scBlocks   = (int)(((long long)E * 32 + T - 1) / T);
  const int wtBlocks   = (int)((MATE + T - 1) / T);

  // One-time: materialize bf16 W^T for all 10 hidden matrices + classifier.
  for (int m2 = 0; m2 < 5; ++m2) {
    k_wt_bf16<<<wtBlocks, T, 0, stream>>>(w_orig + m2 * MATE, wT + m2 * MATE, IN_DIM_, HID_);
    k_wt_bf16<<<wtBlocks, T, 0, stream>>>(w_aug  + m2 * MATE, wT + (5 + m2) * MATE, IN_DIM_, HID_);
  }
  k_wt_bf16<<<(int)((HID_ * OUT_DIM_ + T - 1) / T), T, 0, stream>>>(clf_w, wT + 10 * MATE, HID_, OUT_DIM_);

  for (int br = 0; br < 2; ++br) {
    const long long* rows = (br == 0) ? ei : gei;
    const long long* cols = rows + E;
    const __bf16* Wt  = wT + (size_t)br * 5 * MATE;
    const float*  B   = (br == 0) ? b_orig : b_aug;
    float*        outb = (br == 0) ? out0 : out1;

    k_zero_u32<<<nodeBlocks, T, 0, stream>>>(deg, N);
    k_deg<<<edgeBlocks, T, 0, stream>>>(cols, deg, E);
    k_dinv<<<nodeBlocks, T, 0, stream>>>(deg, dinv, N);

    k_gemm128_acc<<<gemmBlocks, T, 0, stream>>>(x, Wt, B, outb, N, 1, 0);

    const float* src = x;
    float*       dst = txA;
    for (int i = 1; i <= ORDER_; ++i) {
      k_selfloop<<<slBlocks, T, 0, stream>>>(dst, src, dinv, N);
      k_scatter<<<scBlocks, T, 0, stream>>>(rows, cols, dinv, src, dst, E);
      k_gemm128_acc<<<gemmBlocks, T, 0, stream>>>(dst, Wt + (size_t)i * MATE,
                                                  B + (size_t)i * HID_, outb, N, 0,
                                                  (i == ORDER_) ? 1 : 0);
      src = dst;
      dst = (dst == txA) ? txB : txA;
    }
  }

  const long long total = (long long)N * HID_;
  k_fuse<<<(int)((total + T - 1) / T), T, 0, stream>>>(out0, out1, fw, hbuf, total);
  k_clf<<<gemmBlocks, T, 0, stream>>>(hbuf, wT + 10 * MATE, clf_b, (float*)d_out, N);
}
